// st_SNLStage_54391465836931
// MI455X (gfx1250) — compile-verified, matrix operand order checked by reading, not run on hardware
//
#include <hip/hip_runtime.h>
#include <hip/hip_bf16.h>

// ---------------------------------------------------------------------------
// MI455X (gfx1250): bf16 WMMA, f32 accumulate. att kept bf16 (157MB, fits the
// 192MB L2). Dominant att@g GEMM staged into LDS by the Tensor Data Mover.
// ---------------------------------------------------------------------------

typedef __bf16 bf16_t;
typedef __attribute__((ext_vector_type(16))) __bf16 v16bf;
typedef __attribute__((ext_vector_type(8)))  __bf16 v8bf;   // 16 bytes
typedef __attribute__((ext_vector_type(8)))  float  v8f;
typedef __attribute__((ext_vector_type(4)))  unsigned int v4u;
typedef __attribute__((ext_vector_type(8)))  int v8i;
typedef __attribute__((ext_vector_type(4)))  int v4i;

#if defined(__has_builtin)
#if __has_builtin(__builtin_amdgcn_tensor_load_to_lds) && \
    __has_builtin(__builtin_amdgcn_s_wait_tensorcnt)
#define HAVE_TDM 1
#endif
#endif
#ifndef HAVE_TDM
#define HAVE_TDM 0
#endif

#define EPS_BN 1e-5f

static constexpr long kB  = 2;
static constexpr long kC  = 256;
static constexpr long kS  = 6272;   // 8*28*28 = 49*128
static constexpr long kP  = 128;
static constexpr int  kNS = 5;

__device__ __forceinline__ v16bf frag_combine(v8bf lo, v8bf hi) {
  v16bf r;
#pragma unroll
  for (int i = 0; i < 8; ++i) { r[i] = lo[i]; r[i + 8] = hi[i]; }
  return r;
}

__device__ __forceinline__ v8f wmma_bf16(v16bf a, v16bf b, v8f c) {
  return __builtin_amdgcn_wmma_f32_16x16x32_bf16(
      false, a, false, b, (short)0, c, false, false);
}

// ---------------------------------------------------------------------------
// Tiled GEMM: C[M,N] (+)= A[M,K]*B[K,N]. bf16 in, f32 acc.
// A[m*sAM + k*sAK], B[n*sBN + k*sBK]. Block tile 128x128, K-step 32.
// 128 threads = 4 waves (2x2), each wave a 64x64 tile -> 16 WMMA / 8 LDS frags.
// Double-buffered LDS; A optionally staged by the Tensor Data Mover (TDM_A,
// requires sAK==1 row-major A; the TDM pad feature emits the 80B padded rows).
// ---------------------------------------------------------------------------
template <bool OUT_BF16, bool ACC, bool TDM_A>
__global__ __launch_bounds__(128)
void gemm_bf16_wmma(const bf16_t* __restrict__ A, long sAM, long sAK, long sAB,
                    const bf16_t* __restrict__ Bm, long sBN, long sBK, long sBB,
                    float* __restrict__ Cf, bf16_t* __restrict__ Cb,
                    long ldC, long sCB, int K) {
  __shared__ __align__(16) bf16_t Atile[2][128 * 40];  // 80B row pitch
  __shared__ __align__(16) bf16_t Btile[2][128 * 40];

  A  += (long)blockIdx.z * sAB;
  Bm += (long)blockIdx.z * sBB;
  if (OUT_BF16) Cb += (long)blockIdx.z * sCB; else Cf += (long)blockIdx.z * sCB;

  const int tid  = threadIdx.x;
  const int lane = tid & 31;
  const int wid  = tid >> 5;     // 4 waves
  const int wm   = wid >> 1;     // 0..1 -> rows wm*64
  const int wn   = wid & 1;      // 0..1 -> cols wn*64
  const int ln   = lane & 15;
  const int kh   = lane >> 4;

  const long blockRow = (long)blockIdx.x * 128;
  const long blockCol = (long)blockIdx.y * 128;

  v8f acc[4][4];
#pragma unroll
  for (int mi = 0; mi < 4; ++mi)
#pragma unroll
    for (int ni = 0; ni < 4; ++ni) {
      if (ACC) {
#pragma unroll
        for (int j = 0; j < 8; ++j) {
          long row = blockRow + wm * 64 + mi * 16 + j + 8 * kh;
          long col = blockCol + wn * 64 + ni * 16 + ln;
          acc[mi][ni][j] = Cf[row * ldC + col];
        }
      } else {
        acc[mi][ni] = (v8f){};
      }
    }

#if HAVE_TDM
  // D# group1 (constant): data_size=2B, pad 4 DWORDs every 16 DWORDs
  // (64B row -> 80B LDS pitch), tile 32(k) x 128(m), huge tensor dims (no OOB).
  v8i g1;
  if (TDM_A) {
    g1[0] = (int)((1u << 16) | (1u << 20) | (3u << 22) | (3u << 25));
    g1[1] = 0;                              // tensor_dim0[15:0] of 2^30 = 0
    g1[2] = 0x4000;                         // tensor_dim0[31:16]
    g1[3] = (int)(0x4000u | (32u << 16));   // tensor_dim1[31:16] | tile_dim0
    g1[4] = 128;                            // tile_dim1
    g1[5] = (int)(unsigned)sAM;             // tensor_dim0_stride lo32
    g1[6] = 0;
    g1[7] = 0;
  }
#endif

  // ---- staging helpers (register-staged for overlap) ----
  auto loadA = [&](int k0, v8bf* r) {
#pragma unroll
    for (int e = 0; e < 4; ++e) {
      int v = tid + 128 * e;                 // 512 chunks of 8 elems
      if (sAK == 1) {
        int m = v >> 2, kc = (v & 3) * 8;
        r[e] = *(const v8bf*)(A + (blockRow + m) * sAM + (long)(k0 + kc));
      } else {                               // sAM==1: vector along m
        int kc = v >> 4, m0 = (v & 15) * 8;
        r[e] = *(const v8bf*)(A + (long)(k0 + kc) * sAK + (blockRow + m0));
      }
    }
  };
  auto storeA = [&](int buf, const v8bf* r) {
#pragma unroll
    for (int e = 0; e < 4; ++e) {
      int v = tid + 128 * e;
      if (sAK == 1) {
        int m = v >> 2, kc = (v & 3) * 8;
        *(v8bf*)&Atile[buf][m * 40 + kc] = r[e];
      } else {
        int kc = v >> 4, m0 = (v & 15) * 8;
#pragma unroll
        for (int i = 0; i < 8; ++i) Atile[buf][(m0 + i) * 40 + kc] = r[e][i];
      }
    }
  };
  auto loadB = [&](int k0, v8bf* r) {
#pragma unroll
    for (int e = 0; e < 4; ++e) {
      int v = tid + 128 * e;
      if (sBK == 1) {
        int n = v >> 2, kc = (v & 3) * 8;
        r[e] = *(const v8bf*)(Bm + (blockCol + n) * sBN + (long)(k0 + kc));
      } else {                               // sBN==1: vector along n
        int kc = v >> 4, n0 = (v & 15) * 8;
        r[e] = *(const v8bf*)(Bm + (long)(k0 + kc) * sBK + (blockCol + n0));
      }
    }
  };
  auto storeB = [&](int buf, const v8bf* r) {
#pragma unroll
    for (int e = 0; e < 4; ++e) {
      int v = tid + 128 * e;
      if (sBK == 1) {
        int n = v >> 2, kc = (v & 3) * 8;
        *(v8bf*)&Btile[buf][n * 40 + kc] = r[e];
      } else {
        int kc = v >> 4, n0 = (v & 15) * 8;
#pragma unroll
        for (int i = 0; i < 8; ++i) Btile[buf][(n0 + i) * 40 + kc] = r[e][i];
      }
    }
  };
#if HAVE_TDM
  auto issueTdmA = [&](int k0, int buf) {   // wave 0 only; DMA global->LDS
    if (tid < 32) {
      unsigned long long ga =
          (unsigned long long)(const void*)(A + blockRow * sAM + (long)k0);
      unsigned lds = (unsigned)(unsigned long long)(const void*)&Atile[buf][0];
      v4u g0;
      g0[0] = 1u;                                        // count=1
      g0[1] = lds;                                       // lds_addr
      g0[2] = (unsigned)ga;                              // global_addr lo
      g0[3] = (unsigned)(ga >> 32) | (2u << 30);         // global_addr hi|type=2
#if __clang_major__ >= 23
      __builtin_amdgcn_tensor_load_to_lds(g0, g1, (v4i){0, 0, 0, 0},
                                          (v4i){0, 0, 0, 0},
                                          (v8i){0, 0, 0, 0, 0, 0, 0, 0}, 0);
#else
      __builtin_amdgcn_tensor_load_to_lds(g0, g1, (v4i){0, 0, 0, 0},
                                          (v4i){0, 0, 0, 0}, 0);
#endif
    }
  };
#endif

  auto compute = [&](int buf) {
    v16bf af[4], bfr[4];
#pragma unroll
    for (int mi = 0; mi < 4; ++mi) {
      int row = wm * 64 + mi * 16 + ln;
      af[mi] = frag_combine(*(const v8bf*)&Atile[buf][row * 40 + kh * 8],
                            *(const v8bf*)&Atile[buf][row * 40 + kh * 8 + 16]);
    }
#pragma unroll
    for (int ni = 0; ni < 4; ++ni) {
      int col = wn * 64 + ni * 16 + ln;
      bfr[ni] = frag_combine(*(const v8bf*)&Btile[buf][col * 40 + kh * 8],
                             *(const v8bf*)&Btile[buf][col * 40 + kh * 8 + 16]);
    }
#pragma unroll
    for (int mi = 0; mi < 4; ++mi)
#pragma unroll
      for (int ni = 0; ni < 4; ++ni)
        acc[mi][ni] = wmma_bf16(af[mi], bfr[ni], acc[mi][ni]);
  };

  // ---- prologue: fill buffer 0 ----
  {
    v8bf rb[4];
    loadB(0, rb);
#if HAVE_TDM
    if (TDM_A) {
      issueTdmA(0, 0);
      storeB(0, rb);
      if (tid < 32) __builtin_amdgcn_s_wait_tensorcnt(0);
    } else
#endif
    {
      v8bf ra[4];
      loadA(0, ra);
      storeA(0, ra);
      storeB(0, rb);
    }
  }
  __syncthreads();

  // ---- main pipelined loop ----
  int cur = 0;
  for (int k0 = 0; k0 < K; k0 += 32) {
    const int nxt = cur ^ 1;
    const bool more = (k0 + 32) < K;
    v8bf ra[4], rb[4];
    if (more) {
      loadB(k0 + 32, rb);
#if HAVE_TDM
      if (TDM_A) issueTdmA(k0 + 32, nxt);
      else
#endif
        loadA(k0 + 32, ra);
      __builtin_prefetch(Bm + (blockCol + (tid & 127)) * sBN + (long)(k0 + 64) * sBK, 0, 1);
    }
    compute(cur);                 // overlaps with loads / TDM in flight
    if (more) {
#if HAVE_TDM
      if (!TDM_A)
#endif
        storeA(nxt, ra);
      storeB(nxt, rb);
#if HAVE_TDM
      if (TDM_A && tid < 32) __builtin_amdgcn_s_wait_tensorcnt(0);
#endif
    }
    __syncthreads();
    cur = nxt;
  }

  // ---- epilogue ----
#pragma unroll
  for (int mi = 0; mi < 4; ++mi)
#pragma unroll
    for (int ni = 0; ni < 4; ++ni)
#pragma unroll
      for (int j = 0; j < 8; ++j) {
        long row = blockRow + wm * 64 + mi * 16 + j + 8 * kh;
        long col = blockCol + wn * 64 + ni * 16 + ln;
        if (OUT_BF16) Cb[row * ldC + col] = (bf16_t)acc[mi][ni][j];
        else          Cf[row * ldC + col] = acc[mi][ni][j];
      }
}

// ---------------------------------------------------------------------------
// Fused attention softmax: att = softmax_rows(t @ p) in bf16.
// Wave keeps 16x128 of t resident in VGPRs; streams 392 pT column tiles.
// Online (m,s) kept PER LANE (each lane only sees its own column slots), so
// the hot loop has zero cross-lane ops; one 4-step pair-merge at the end.
// ---------------------------------------------------------------------------
__global__ __launch_bounds__(256)
void attn_softmax(const bf16_t* __restrict__ t, const bf16_t* __restrict__ pT,
                  bf16_t* __restrict__ att, int S) {
  const long b = blockIdx.z;
  t   += b * (long)S * kP;
  pT  += b * (long)S * kP;
  att += b * (long)S * S;

  const int lane = threadIdx.x & 31;
  const int wid  = threadIdx.x >> 5;
  const int ln   = lane & 15;
  const int kh   = lane >> 4;
  const int rowbase = blockIdx.x * 128 + wid * 16;

  v16bf af[4];
#pragma unroll
  for (int kk = 0; kk < 4; ++kk) {
    long row = rowbase + ln;
    int  kb  = kk * 32 + kh * 8;
    af[kk] = frag_combine(*(const v8bf*)(t + row * kP + kb),
                          *(const v8bf*)(t + row * kP + kb + 16));
  }

  float m[8], s[8];
#pragma unroll
  for (int j = 0; j < 8; ++j) { m[j] = -3.0e38f; s[j] = 0.0f; }

  const int ntiles = S / 16;
  // ---- pass 1: per-lane online max/sum ----
  for (int ct = 0; ct < ntiles; ++ct) {
    const int colbase = ct * 16;
    v8f accv = (v8f){};
#pragma unroll
    for (int kk = 0; kk < 4; ++kk) {
      long n  = colbase + ln;
      int  kb = kk * 32 + kh * 8;
      v16bf bfr = frag_combine(*(const v8bf*)(pT + n * kP + kb),
                               *(const v8bf*)(pT + n * kP + kb + 16));
      accv = wmma_bf16(af[kk], bfr, accv);
    }
#pragma unroll
    for (int j = 0; j < 8; ++j) {
      float v  = accv[j];
      float mn = fmaxf(m[j], v);
      s[j] = s[j] * __expf(m[j] - mn) + __expf(v - mn);
      m[j] = mn;
    }
  }
  // ---- merge (m,s) across the 16 lanes of each half-wave ----
#pragma unroll
  for (int j = 0; j < 8; ++j) {
#pragma unroll
    for (int off = 8; off > 0; off >>= 1) {
      float om = __shfl_xor(m[j], off, 16);
      float os = __shfl_xor(s[j], off, 16);
      float mn = fmaxf(m[j], om);
      s[j] = s[j] * __expf(m[j] - mn) + os * __expf(om - mn);
      m[j] = mn;
    }
  }
  float inv[8];
#pragma unroll
  for (int j = 0; j < 8; ++j) inv[j] = 1.0f / s[j];

  // ---- pass 2: recompute logits, store exp(l-m)/s ----
  for (int ct = 0; ct < ntiles; ++ct) {
    const int colbase = ct * 16;
    v8f accv = (v8f){};
#pragma unroll
    for (int kk = 0; kk < 4; ++kk) {
      long n  = colbase + ln;
      int  kb = kk * 32 + kh * 8;
      v16bf bfr = frag_combine(*(const v8bf*)(pT + n * kP + kb),
                               *(const v8bf*)(pT + n * kP + kb + 16));
      accv = wmma_bf16(af[kk], bfr, accv);
    }
#pragma unroll
    for (int j = 0; j < 8; ++j) {
      float e = __expf(accv[j] - m[j]) * inv[j];
      long row = rowbase + j + 8 * kh;
      long col = colbase + ln;
      att[row * (long)S + col] = (bf16_t)e;
    }
  }
}

// ---------------------------------------------------------------------------
__global__ __launch_bounds__(256)
void bn_stats(const float* __restrict__ y, float* __restrict__ mu,
              float* __restrict__ rvar, long S, long CS) {
  const int c = blockIdx.x, tid = threadIdx.x;
  float sum = 0.f, sq = 0.f;
  for (long i = tid; i < S; i += 256) {
    float a = y[(long)c * S + i];
    float b = y[CS + (long)c * S + i];
    sum += a + b;
    sq  += a * a + b * b;
  }
  __shared__ float rs[256], rq[256];
  rs[tid] = sum; rq[tid] = sq;
  __syncthreads();
  for (int off = 128; off > 0; off >>= 1) {
    if (tid < off) { rs[tid] += rs[tid + off]; rq[tid] += rq[tid + off]; }
    __syncthreads();
  }
  if (tid == 0) {
    float n   = (float)(2 * S);
    float muv = rs[0] / n;
    float var = rq[0] / n - muv * muv;
    mu[c]   = muv;
    rvar[c] = rsqrtf(var + EPS_BN);
  }
}

__global__ __launch_bounds__(256)
void bn_apply(const float* __restrict__ y, const float* __restrict__ mu,
              const float* __restrict__ rvar, const float* __restrict__ gamma,
              const float* __restrict__ beta, float* __restrict__ outf,
              bf16_t* __restrict__ outb, long S, long CS) {
  long idx = (long)blockIdx.x * 256 + threadIdx.x;
  long c   = (idx % CS) / S;
  float v  = (y[idx] - mu[c]) * rvar[c] * gamma[c] + beta[c] + outf[idx];
  outf[idx] = v;
  outb[idx] = (bf16_t)v;
}

__global__ __launch_bounds__(256)
void cvt_f32_bf16(const float* __restrict__ src, bf16_t* __restrict__ dst, long n) {
  long i = (long)blockIdx.x * 256 + threadIdx.x;
  if (i < n) dst[i] = (bf16_t)src[i];
}

__global__ __launch_bounds__(256)
void init_out(const float* __restrict__ x, float* __restrict__ outf,
              bf16_t* __restrict__ outb, long n) {
  long i = (long)blockIdx.x * 256 + threadIdx.x;
  if (i < n) { float v = x[i]; outf[i] = v; outb[i] = (bf16_t)v; }
}

// ---------------------------------------------------------------------------
extern "C" void kernel_launch(void* const* d_in, const int* in_sizes, int n_in,
                              void* d_out, int out_size, void* d_ws, size_t ws_size,
                              hipStream_t stream) {
  const float* x     = (const float*)d_in[0];
  const float* Wt    = (const float*)d_in[1];
  const float* Wp    = (const float*)d_in[2];
  const float* Wg    = (const float*)d_in[3];
  const float* W1    = (const float*)d_in[4];
  const float* W2    = (const float*)d_in[5];
  const float* gamma = (const float*)d_in[6];
  const float* beta  = (const float*)d_in[7];
  float* outf = (float*)d_out;

  char* base = (char*)d_ws;
  size_t off = 0;
  auto carve = [&](size_t bytes) -> void* {
    off = (off + 255) & ~(size_t)255;
    void* p = base + off;
    off += bytes;
    return p;
  };
  bf16_t* att  = (bf16_t*)carve(kB * kS * kS * sizeof(bf16_t));  // L2-resident
  bf16_t* outb = (bf16_t*)carve(kB * kC * kS * sizeof(bf16_t));
  bf16_t* tb   = (bf16_t*)carve(kB * kS * kP * sizeof(bf16_t));
  bf16_t* pTb  = (bf16_t*)carve(kB * kS * kP * sizeof(bf16_t));
  bf16_t* gb   = (bf16_t*)carve(kB * kS * kP * sizeof(bf16_t));
  bf16_t* x2b  = (bf16_t*)carve(kB * kS * kP * sizeof(bf16_t));
  float*  y    = (float*)carve(kB * kC * kS * sizeof(float));
  float*  mu   = (float*)carve(kC * sizeof(float));
  float*  rv   = (float*)carve(kC * sizeof(float));
  bf16_t* WtB  = (bf16_t*)carve(kP * kC * sizeof(bf16_t));
  bf16_t* WpB  = (bf16_t*)carve(kP * kC * sizeof(bf16_t));
  bf16_t* WgB  = (bf16_t*)carve(kNS * kP * kC * sizeof(bf16_t));
  bf16_t* W1B  = (bf16_t*)carve(kNS * kC * kP * sizeof(bf16_t));
  bf16_t* W2B  = (bf16_t*)carve(kNS * kC * kP * sizeof(bf16_t));
  (void)ws_size; (void)n_in; (void)in_sizes; (void)out_size;

  const long CS = kC * kS;

  cvt_f32_bf16<<<dim3((kP * kC + 255) / 256), 256, 0, stream>>>(Wt, WtB, kP * kC);
  cvt_f32_bf16<<<dim3((kP * kC + 255) / 256), 256, 0, stream>>>(Wp, WpB, kP * kC);
  cvt_f32_bf16<<<dim3((kNS * kP * kC + 255) / 256), 256, 0, stream>>>(Wg, WgB, kNS * kP * kC);
  cvt_f32_bf16<<<dim3((kNS * kC * kP + 255) / 256), 256, 0, stream>>>(W1, W1B, kNS * kC * kP);
  cvt_f32_bf16<<<dim3((kNS * kC * kP + 255) / 256), 256, 0, stream>>>(W2, W2B, kNS * kC * kP);
  init_out<<<dim3((kB * CS + 255) / 256), 256, 0, stream>>>(x, outf, outb, kB * CS);

  // projections t / pT = xf^T @ W^T : [S,P], K=C
  gemm_bf16_wmma<true, false, false><<<dim3(kS / 128, kP / 128, kB), 128, 0, stream>>>(
      outb, 1, kS, CS, WtB, kC, 1, 0, nullptr, tb, kP, kS * kP, (int)kC);
  gemm_bf16_wmma<true, false, false><<<dim3(kS / 128, kP / 128, kB), 128, 0, stream>>>(
      outb, 1, kS, CS, WpB, kC, 1, 0, nullptr, pTb, kP, kS * kP, (int)kC);

  attn_softmax<<<dim3(kS / 128, 1, kB), 256, 0, stream>>>(tb, pTb, att, (int)kS);

  for (int i = 0; i < kNS; ++i) {
    // g = out^T @ Wg[i]^T : [S,P]
    gemm_bf16_wmma<true, false, false><<<dim3(kS / 128, kP / 128, kB), 128, 0, stream>>>(
        outb, 1, kS, CS, WgB + (long)i * kP * kC, kC, 1, 0,
        nullptr, gb, kP, kS * kP, (int)kC);
    // x1 = W1[i] @ g^T : [C,S]
    gemm_bf16_wmma<false, false, false><<<dim3(kC / 128, kS / 128, kB), 128, 0, stream>>>(
        W1B + (long)i * kC * kP, kP, 1, 0, gb, kP, 1, kS * kP,
        y, nullptr, kS, CS, (int)kP);
    // x2 = att @ g : [S,P], K=S — TDM streams the 157MB bf16 att from L2
    gemm_bf16_wmma<true, false, true><<<dim3(kS / 128, kP / 128, kB), 128, 0, stream>>>(
        att, kS, 1, kS * kS, gb, 1, kP, kS * kP,
        nullptr, x2b, kP, kS * kP, (int)kS);
    // y += W2[i] @ x2^T (fused x1 + x2)
    gemm_bf16_wmma<false, true, false><<<dim3(kC / 128, kS / 128, kB), 128, 0, stream>>>(
        W2B + (long)i * kC * kP, kP, 1, 0, x2b, kP, 1, kS * kP,
        y, nullptr, kS, CS, (int)kP);
    bn_stats<<<dim3((unsigned)kC), 256, 0, stream>>>(y, mu, rv, kS, CS);
    bn_apply<<<dim3((unsigned)(kB * CS / 256)), 256, 0, stream>>>(
        y, mu, rv, gamma + (long)i * kC, beta + (long)i * kC, outf, outb, kS, CS);
  }
}
